// MultiSourceTransformer_88192858456753
// MI455X (gfx1250) — matrix operand order repaired and checked by reading, not compiled, not run on hardware
//
#include <hip/hip_runtime.h>
#include <hip/hip_bf16.h>
#include <math.h>

// Problem constants (from reference)
#define B_     4
#define T_     512
#define M_     8
#define DIM_   128
#define HEADS_ 8
#define DHEAD_ 64
#define DEPTH_ 4
#define INNER_ 512
#define R_     (B_ * T_)   // 2048 rows per source

typedef __attribute__((ext_vector_type(16))) __bf16 v16bf;
typedef __attribute__((ext_vector_type(8)))  __bf16 v8bf;
typedef __attribute__((ext_vector_type(8)))  float  v8f;

__device__ __forceinline__ unsigned short f32_to_bf16(float f) {
  union { float f; unsigned u; } v; v.f = f;
  unsigned r = v.u + 0x7FFFu + ((v.u >> 16) & 1u);   // round-to-nearest-even
  return (unsigned short)(r >> 16);
}

// Assemble a v16bf WMMA operand from two contiguous 16-byte chunks.
__device__ __forceinline__ v16bf load_frag(const unsigned short* p0,
                                           const unsigned short* p1) {
  v8bf lo = *(const v8bf*)p0;
  v8bf hi = *(const v8bf*)p1;
  return __builtin_shufflevector(lo, hi, 0,1,2,3,4,5,6,7,8,9,10,11,12,13,14,15);
}

__device__ __forceinline__ v8f wmma_bf16(v16bf a, v16bf b, v8f c) {
  return __builtin_amdgcn_wmma_f32_16x16x32_bf16(false, a, false, b, (short)0, c,
                                                 false, false);
}

// gfx1250 async copy: global -> LDS, 16 bytes per lane, tracked by ASYNCcnt.
// lds_off is the 32-bit LDS byte offset (low word of the generic address).
__device__ __forceinline__ void async_g2l_b128(unsigned lds_off,
                                               const void* gaddr) {
  asm volatile("global_load_async_to_lds_b128 %0, %1, off"
               :: "v"(lds_off), "v"(gaddr) : "memory");
}
__device__ __forceinline__ void async_g2l_b128_off16(unsigned lds_off,
                                                     const void* gaddr) {
  // INST_OFFSET applies to both the LDS and global addresses (ISA 10.7).
  asm volatile("global_load_async_to_lds_b128 %0, %1, off offset:16"
               :: "v"(lds_off), "v"(gaddr) : "memory");
}
__device__ __forceinline__ void wait_async0() {
  asm volatile("s_wait_asynccnt 0" ::: "memory");
}
__device__ __forceinline__ unsigned lds_offset(const void* p) {
  // Generic LDS addresses carry the byte offset in addr[31:0] (aperture in
  // the high word), so truncation yields the LDS offset.
  return (unsigned)(unsigned long long)p;
}

// ---------------------------------------------------------------------------
// Weight transpose + cast: src (S, K, N) f32  ->  dst (S, N, K) bf16
// ---------------------------------------------------------------------------
__global__ void transpose_cast_kernel(const float* __restrict__ src,
                                      unsigned short* __restrict__ dst,
                                      int K, int N, int S) {
  size_t idx   = (size_t)blockIdx.x * blockDim.x + threadIdx.x;
  size_t per   = (size_t)K * N;
  size_t total = per * (size_t)S;
  if (idx >= total) return;
  int    s   = (int)(idx / per);
  size_t rem = idx % per;
  int    n   = (int)(rem / K);
  int    k   = (int)(rem % K);
  dst[idx] = f32_to_bf16(src[(size_t)s * per + (size_t)k * N + n]);
}

// ---------------------------------------------------------------------------
// LayerNorm over DIM=128 + cast to bf16.  x: (R, M, 128) f32 -> xn: (M, R, 128)
// ---------------------------------------------------------------------------
__global__ __launch_bounds__(128) void ln_cast_kernel(
    const float* __restrict__ x, const float* __restrict__ g,
    const float* __restrict__ bta, unsigned short* __restrict__ xn) {
  int rm  = blockIdx.x;            // r * M + m
  int r   = rm / M_;
  int m   = rm % M_;
  int tid = threadIdx.x;
  __shared__ float red[128];
  float v = x[((size_t)r * M_ + m) * DIM_ + tid];
  red[tid] = v; __syncthreads();
  for (int s = 64; s > 0; s >>= 1) { if (tid < s) red[tid] += red[tid + s]; __syncthreads(); }
  float mu = red[0] * (1.0f / DIM_); __syncthreads();
  float d = v - mu;
  red[tid] = d * d; __syncthreads();
  for (int s = 64; s > 0; s >>= 1) { if (tid < s) red[tid] += red[tid + s]; __syncthreads(); }
  float var = red[0] * (1.0f / DIM_);
  float o = d * rsqrtf(var + 1e-5f) * g[tid] + bta[tid];
  xn[((size_t)m * R_ + r) * DIM_ + tid] = f32_to_bf16(o);
}

// ---------------------------------------------------------------------------
// WMMA GEMM with double-buffered async LDS staging.
//   A  : bf16 (M_, R, K) row-major;  Bt : bf16 (M_, N, K) (pre-transposed)
// Block = 256 threads (8 waves); tile 128 rows x 64 cols; K step 32.
// Tiles for step k+1 stream into LDS via GLOBAL_LOAD_ASYNC_TO_LDS_B128 while
// the 4 WMMAs of step k execute; s_wait_asynccnt 0 + barrier at step end.
// EPI 0: scatter q/k/vT bf16 (qkv).  EPI 1: f32 store (M_, R, N).
// EPI 2: residual accumulate into x (R, M_, DIM) with optional bias (M_, N).
// ---------------------------------------------------------------------------
template <int EPI>
__global__ __launch_bounds__(256) void gemm_wmma(
    const unsigned short* __restrict__ A,
    const unsigned short* __restrict__ Bt,
    int K, int N,
    float* __restrict__ outF,
    const float* __restrict__ bias,
    unsigned short* __restrict__ qd,
    unsigned short* __restrict__ kd,
    unsigned short* __restrict__ vTd,
    float qscale) {
  const int m    = blockIdx.z;
  const int rb   = blockIdx.y * 128;
  const int nb   = blockIdx.x * 64;
  const int tid  = threadIdx.x;
  const int wave = tid >> 5;
  const int lane = tid & 31;
  const int ln16 = lane & 15;
  const int hi   = lane >> 4;

  __shared__ __align__(16) unsigned short ldsA[2][128 * 40];  // padded stride 40
  __shared__ __align__(16) unsigned short ldsB[2][64 * 40];

  const unsigned short* Am = A  + (size_t)m * R_ * K;
  const unsigned short* Bm = Bt + (size_t)m * N  * K;

  // Per-thread staging assignment: A = 32B (two b128), B = 16B (one b128).
  const int ar = tid >> 1;            // 0..127
  const int ac = (tid & 1) * 16;      // 0 or 16 halfs
  const int bn = tid >> 2;            // 0..63
  const int bc = (tid & 3) * 8;       // 0,8,16,24 halfs
  const unsigned short* aSrc = Am + (size_t)(rb + ar) * K + ac;
  const unsigned short* bSrc = Bm + (size_t)(nb + bn) * K + bc;
  unsigned aDst[2], bDst[2];
  aDst[0] = lds_offset(&ldsA[0][ar * 40 + ac]);
  aDst[1] = lds_offset(&ldsA[1][ar * 40 + ac]);
  bDst[0] = lds_offset(&ldsB[0][bn * 40 + bc]);
  bDst[1] = lds_offset(&ldsB[1][bn * 40 + bc]);

  v8f acc[4] = {};

  // Prologue: stage tile 0.
  async_g2l_b128(aDst[0], aSrc);
  async_g2l_b128_off16(aDst[0], aSrc);
  async_g2l_b128(bDst[0], bSrc);
  wait_async0();
  __syncthreads();

  int buf = 0;
  for (int k0 = 0; k0 < K; k0 += 32) {
    // Kick off async staging of the next K-tile into the other buffer.
    if (k0 + 32 < K) {
      const unsigned short* as = aSrc + (k0 + 32);
      const unsigned short* bs = bSrc + (k0 + 32);
      async_g2l_b128(aDst[buf ^ 1], as);
      async_g2l_b128_off16(aDst[buf ^ 1], as);
      async_g2l_b128(bDst[buf ^ 1], bs);
    }

    // A fragment (16x32): lane row = ln16, K = 16*hi + j (contiguous).
    const unsigned short* arow = &ldsA[buf][(wave * 16 + ln16) * 40];
    v16bf afrag = load_frag(arow + 16 * hi, arow + 16 * hi + 8);
#pragma unroll
    for (int t = 0; t < 4; ++t) {
      // B fragment (32x16): lane col = ln16, K = 16*hi + j (contiguous).
      const unsigned short* brow = &ldsB[buf][(t * 16 + ln16) * 40 + 16 * hi];
      v16bf bfrag = load_frag(brow, brow + 8);
      acc[t] = wmma_bf16(afrag, bfrag, acc[t]);
    }

    wait_async0();
    __syncthreads();
    buf ^= 1;
  }

  // Epilogue.  C/D layout: lane holds (row = i + 8*hi, col = ln16) for i=0..7.
  if (EPI == 0) {
#pragma unroll
    for (int t = 0; t < 4; ++t) {
      int ng   = nb + t * 16 + ln16;
      int part = ng >> 9;          // 0 = q, 1 = k, 2 = v
      int oo   = ng & 511;
      int h    = oo >> 6;
      int d    = oo & 63;
#pragma unroll
      for (int i = 0; i < 8; ++i) {
        int   rg = rb + wave * 16 + i + 8 * hi;
        float v  = acc[t][i];
        if (part == 0)
          qd[(((size_t)h * M_ + m) * R_ + rg) * 64 + d] = f32_to_bf16(v * qscale);
        else if (part == 1)
          kd[(((size_t)h * M_ + m) * R_ + rg) * 64 + d] = f32_to_bf16(v);
        else  // V stored transposed (h, m, d, r) so P*V B-fragments are contiguous
          vTd[(((size_t)h * M_ + m) * 64 + d) * R_ + rg] = f32_to_bf16(v);
      }
    }
  } else if (EPI == 1) {
#pragma unroll
    for (int t = 0; t < 4; ++t) {
      int ng = nb + t * 16 + ln16;
#pragma unroll
      for (int i = 0; i < 8; ++i) {
        int rg = rb + wave * 16 + i + 8 * hi;
        outF[((size_t)m * R_ + rg) * N + ng] = acc[t][i];
      }
    }
  } else {  // EPI == 2: residual accumulate into x (R, M, DIM), N == DIM_
#pragma unroll
    for (int t = 0; t < 4; ++t) {
      int   ng = nb + t * 16 + ln16;
      float bv = bias ? bias[(size_t)m * N + ng] : 0.0f;
#pragma unroll
      for (int i = 0; i < 8; ++i) {
        int    rg = rb + wave * 16 + i + 8 * hi;
        size_t xi = ((size_t)rg * M_ + m) * DIM_ + ng;
        outF[xi] += acc[t][i] + bv;
      }
    }
  }
}

// ---------------------------------------------------------------------------
// Flash attention per (b, h, m).  Q/K: (H, M, R, 64) bf16 (q pre-scaled),
// V: (H, M, 64, R) bf16 (transposed).  Each wave owns 16 query rows with
// online softmax; grid = (B*H*M, T/128), 8 waves per block.
// o out: (M, R, INNER) bf16 with inner = h*64 + d  (ready as Wout GEMM A).
// ---------------------------------------------------------------------------
__global__ __launch_bounds__(256) void attn_kernel(
    const unsigned short* __restrict__ q,
    const unsigned short* __restrict__ kk,
    const unsigned short* __restrict__ vT,
    const unsigned char* __restrict__ mask,   // (B, T, M), nonzero = masked
    unsigned short* __restrict__ o) {
  int bidx = blockIdx.x;
  int m = bidx % M_;
  int h = (bidx / M_) % HEADS_;
  int b = bidx / (M_ * HEADS_);
  int wave = threadIdx.x >> 5;
  int lane = threadIdx.x & 31;
  int ln16 = lane & 15;
  int hi   = lane >> 4;

  const size_t hm = (size_t)h * M_ + m;
  const unsigned short* Q  = q  + hm * R_ * 64;
  const unsigned short* Kp = kk + hm * R_ * 64;
  const unsigned short* Vp = vT + hm * 64 * R_;
  const int qrow0 = b * T_ + blockIdx.y * 128 + wave * 16;

  __shared__ __align__(16) unsigned short ldsP[8][16 * 40];  // per-wave P bounce
  unsigned short* myP = &ldsP[wave][0];

  // Q fragments: two 16x32 A-frags over d (=wmma K dim).
  v16bf qf[2];
#pragma unroll
  for (int c = 0; c < 2; ++c) {
    const unsigned short* base = Q + (size_t)(qrow0 + ln16) * 64 + 32 * c;
    qf[c] = load_frag(base + 16 * hi, base + 16 * hi + 8);
  }

  v8f   accO[4] = {};
  float mrow[8], lrow[8];
#pragma unroll
  for (int i = 0; i < 8; ++i) { mrow[i] = -INFINITY; lrow[i] = 0.0f; }

  for (int j0 = 0; j0 < T_; j0 += 32) {
    // Prefetch next key block: one K row and two V rows per lane
    // (global_prefetch_b8 pulls whole cachelines).
    if (j0 + 32 < T_) {
      int nk = b * T_ + j0 + 32;
      __builtin_prefetch(Kp + (size_t)(nk + lane) * 64, 0, 0);
      __builtin_prefetch(Vp + (size_t)(lane) * R_ + nk, 0, 0);
      __builtin_prefetch(Vp + (size_t)(32 + lane) * R_ + nk, 0, 0);
    }
    // S = Q * K^T for 32 keys (2 tiles of 16).
    v8f s[2] = {};
#pragma unroll
    for (int t = 0; t < 2; ++t) {
      int key = b * T_ + j0 + t * 16 + ln16;  // B-frag column = key index
#pragma unroll
      for (int c = 0; c < 2; ++c) {
        const unsigned short* kb = Kp + (size_t)key * 64 + 32 * c + 16 * hi;
        v16bf kf = load_frag(kb, kb + 8);
        s[t] = wmma_bf16(qf[c], kf, s[t]);
      }
    }
    // Key-padding mask (same for all query rows).
#pragma unroll
    for (int t = 0; t < 2; ++t) {
      int tj = j0 + t * 16 + ln16;
      if (mask[((size_t)b * T_ + tj) * M_ + m]) {
#pragma unroll
        for (int i = 0; i < 8; ++i) s[t][i] = -INFINITY;
      }
    }
    // Online softmax: row reductions across the 16-lane half (xor 1,2,4,8).
    float p0[8], p1[8];
#pragma unroll
    for (int i = 0; i < 8; ++i) {
      float mx = fmaxf(s[0][i], s[1][i]);
#pragma unroll
      for (int x = 1; x <= 8; x <<= 1) mx = fmaxf(mx, __shfl_xor(mx, x, 32));
      float nm = fmaxf(mrow[i], mx);
      p0[i] = __expf(s[0][i] - nm);
      p1[i] = __expf(s[1][i] - nm);
      float rs = p0[i] + p1[i];
#pragma unroll
      for (int x = 1; x <= 8; x <<= 1) rs += __shfl_xor(rs, x, 32);
      float fac = __expf(mrow[i] - nm);
      lrow[i] = lrow[i] * fac + rs;
      mrow[i] = nm;
#pragma unroll
      for (int t = 0; t < 4; ++t) accO[t][i] *= fac;
    }
    // Bounce P through per-wave LDS: C-layout -> A-layout (plain LDS
    // stores and loads, wave-local).
#pragma unroll
    for (int i = 0; i < 8; ++i) {
      myP[(i + 8 * hi) * 40 + ln16]      = f32_to_bf16(p0[i]);
      myP[(i + 8 * hi) * 40 + 16 + ln16] = f32_to_bf16(p1[i]);
    }
    __threadfence_block();
    const unsigned short* prow = myP + ln16 * 40;
    v16bf pf = load_frag(prow + 16 * hi, prow + 16 * hi + 8);
    // O += P (16x32) * V (32 keys x 64 d): 4 WMMA, V^T rows are contiguous.
#pragma unroll
    for (int t = 0; t < 4; ++t) {
      const unsigned short* vb = Vp + (size_t)(t * 16 + ln16) * R_ + b * T_ + j0 + 16 * hi;
      v16bf vf = load_frag(vb, vb + 8);
      accO[t] = wmma_bf16(pf, vf, accO[t]);
    }
    __threadfence_block();
  }

  // Normalize and emit o in Wout-GEMM A layout (M, R, INNER).
#pragma unroll
  for (int i = 0; i < 8; ++i) {
    float inv = 1.0f / lrow[i];
    int   rg  = qrow0 + i + 8 * hi;
#pragma unroll
    for (int t = 0; t < 4; ++t) {
      int d = t * 16 + ln16;
      o[((size_t)m * R_ + rg) * INNER_ + h * 64 + d] =
          f32_to_bf16(accO[t][i] * inv);
    }
  }
}

// ---------------------------------------------------------------------------
// GEGLU: hgl (M, R, 1024) f32 + b1 (M, 1024) -> hb (M, R, 512) bf16
// ---------------------------------------------------------------------------
__global__ void geglu_kernel(const float* __restrict__ hgl,
                             const float* __restrict__ b1,
                             unsigned short* __restrict__ hb) {
  size_t idx   = (size_t)blockIdx.x * blockDim.x + threadIdx.x;
  size_t total = (size_t)M_ * R_ * 512;
  if (idx >= total) return;
  int    n  = (int)(idx & 511);
  size_t mr = idx >> 9;
  int    m  = (int)(mr / R_);
  const float* row = hgl + mr * 1024;
  float hv = row[n]       + b1[(size_t)m * 1024 + n];
  float gv = row[512 + n] + b1[(size_t)m * 1024 + 512 + n];
  float ge = 0.5f * gv * (1.0f + erff(gv * 0.70710678118654752f));
  hb[idx]  = f32_to_bf16(hv * ge);
}

// ---------------------------------------------------------------------------
extern "C" void kernel_launch(void* const* d_in, const int* in_sizes, int n_in,
                              void* d_out, int out_size, void* d_ws, size_t ws_size,
                              hipStream_t stream) {
  (void)in_sizes; (void)n_in; (void)ws_size;
  const float*         x_in = (const float*)d_in[0];
  const unsigned char* mask = (const unsigned char*)d_in[1];
  // params dict order: ln1_g, ln1_b, Wqkv, Wout, ln2_g, ln2_b, W1, b1, W2, b2
  const float* ln1_g = (const float*)d_in[2];
  const float* ln1_b = (const float*)d_in[3];
  const float* Wqkv  = (const float*)d_in[4];
  const float* Wout  = (const float*)d_in[5];
  const float* ln2_g = (const float*)d_in[6];
  const float* ln2_b = (const float*)d_in[7];
  const float* W1    = (const float*)d_in[8];
  const float* b1    = (const float*)d_in[9];
  const float* W2    = (const float*)d_in[10];
  const float* b2    = (const float*)d_in[11];

  // Workspace carve (256-byte aligned chunks).
  char* ws = (char*)d_ws;
  auto alloc = [&](size_t bytes) -> void* {
    void* p = (void*)ws;
    ws += (bytes + 255) & ~(size_t)255;
    return p;
  };
  unsigned short* WqkvT = (unsigned short*)alloc(sizeof(short) * DEPTH_ * M_ * 1536 * 128);
  unsigned short* WoutT = (unsigned short*)alloc(sizeof(short) * DEPTH_ * M_ * 128 * 512);
  unsigned short* W1T   = (unsigned short*)alloc(sizeof(short) * DEPTH_ * M_ * 1024 * 128);
  unsigned short* W2T   = (unsigned short*)alloc(sizeof(short) * DEPTH_ * M_ * 128 * 512);
  unsigned short* xn    = (unsigned short*)alloc(sizeof(short) * M_ * R_ * DIM_);
  unsigned short* qb    = (unsigned short*)alloc(sizeof(short) * HEADS_ * M_ * R_ * DHEAD_);
  unsigned short* kb    = (unsigned short*)alloc(sizeof(short) * HEADS_ * M_ * R_ * DHEAD_);
  unsigned short* vb    = (unsigned short*)alloc(sizeof(short) * HEADS_ * M_ * R_ * DHEAD_);
  unsigned short* ob    = (unsigned short*)alloc(sizeof(short) * M_ * R_ * INNER_);
  float*          hgl   = (float*)alloc(sizeof(float) * M_ * R_ * 1024);
  unsigned short* hb    = (unsigned short*)alloc(sizeof(short) * M_ * R_ * 512);

  // Working x lives in d_out (residual stream, updated in place).
  hipMemcpyAsync(d_out, x_in, sizeof(float) * (size_t)out_size,
                 hipMemcpyDeviceToDevice, stream);
  float* xw = (float*)d_out;

  // Convert + transpose all weights to bf16 (N, K) per (layer, source).
  {
    size_t tq = (size_t)DEPTH_ * M_ * 128 * 1536;
    transpose_cast_kernel<<<(tq + 255) / 256, 256, 0, stream>>>(Wqkv, WqkvT, 128, 1536, DEPTH_ * M_);
    size_t to = (size_t)DEPTH_ * M_ * 512 * 128;
    transpose_cast_kernel<<<(to + 255) / 256, 256, 0, stream>>>(Wout, WoutT, 512, 128, DEPTH_ * M_);
    size_t t1 = (size_t)DEPTH_ * M_ * 128 * 1024;
    transpose_cast_kernel<<<(t1 + 255) / 256, 256, 0, stream>>>(W1, W1T, 128, 1024, DEPTH_ * M_);
    size_t t2 = (size_t)DEPTH_ * M_ * 512 * 128;
    transpose_cast_kernel<<<(t2 + 255) / 256, 256, 0, stream>>>(W2, W2T, 512, 128, DEPTH_ * M_);
  }

  const float qscale = 0.125f;  // DHEAD^-0.5

  for (int l = 0; l < DEPTH_; ++l) {
    // ---- attention block ----
    ln_cast_kernel<<<R_ * M_, 128, 0, stream>>>(xw, ln1_g + (size_t)l * DIM_,
                                                ln1_b + (size_t)l * DIM_, xn);
    {
      dim3 g(1536 / 64, R_ / 128, M_);
      gemm_wmma<0><<<g, 256, 0, stream>>>(
          xn, WqkvT + (size_t)l * M_ * 1536 * 128, 128, 1536,
          nullptr, nullptr, qb, kb, vb, qscale);
    }
    {
      dim3 ga(B_ * HEADS_ * M_, T_ / 128);
      attn_kernel<<<ga, 256, 0, stream>>>(qb, kb, vb, mask, ob);
    }
    {
      dim3 g(DIM_ / 64, R_ / 128, M_);
      gemm_wmma<2><<<g, 256, 0, stream>>>(
          ob, WoutT + (size_t)l * M_ * 128 * 512, 512, DIM_,
          xw, nullptr, nullptr, nullptr, nullptr, 1.0f);
    }
    // ---- feed-forward block ----
    ln_cast_kernel<<<R_ * M_, 128, 0, stream>>>(xw, ln2_g + (size_t)l * DIM_,
                                                ln2_b + (size_t)l * DIM_, xn);
    {
      dim3 g(1024 / 64, R_ / 128, M_);
      gemm_wmma<1><<<g, 256, 0, stream>>>(
          xn, W1T + (size_t)l * M_ * 1024 * 128, 128, 1024,
          hgl, nullptr, nullptr, nullptr, nullptr, 1.0f);
    }
    {
      size_t tg = (size_t)M_ * R_ * 512;
      geglu_kernel<<<(tg + 255) / 256, 256, 0, stream>>>(
          hgl, b1 + (size_t)l * M_ * 1024, hb);
    }
    {
      dim3 g(DIM_ / 64, R_ / 128, M_);
      gemm_wmma<2><<<g, 256, 0, stream>>>(
          hb, W2T + (size_t)l * M_ * 128 * 512, 512, DIM_,
          xw, b2 + (size_t)l * M_ * DIM_, nullptr, nullptr, nullptr, 1.0f);
    }
  }
}
